// Autoformer_21131239096329
// MI455X (gfx1250) — compile-verified, matrix operand order checked by reading, not run on hardware
//
#include <hip/hip_runtime.h>
#include <cstdint>

// ---------------- problem constants ----------------
#define BSZ   16
#define LSEQ  2048
#define DMOD  1024
#define TOPK  7            // int(log(2048)) == 7
#define LMASK (LSEQ - 1)

typedef __attribute__((ext_vector_type(16))) _Float16 v16h;
typedef __attribute__((ext_vector_type(8)))  _Float16 v8h;
typedef __attribute__((ext_vector_type(4)))  _Float16 v4h;
typedef __attribute__((ext_vector_type(8)))  float    v8f;

#define BK    32
#define LDSW  40   // LDS row stride in halves: 32 data + 8 pad (= 80B, multiple of 16B)

// ---- gfx1250 async global->LDS copy (ASYNCcnt), CDNA5 ISA ----
__device__ __forceinline__ void async_copy_b128(const void* gptr, const void* lptr) {
  // low 32 bits of the flat shared-pointer value are the LDS offset
  uint32_t loff = (uint32_t)(uint64_t)lptr;
  asm volatile("global_load_async_to_lds_b128 %0, %1, off"
               :: "v"(loff), "v"(gptr) : "memory");
}
__device__ __forceinline__ void wait_async0() {
  asm volatile("s_wait_asynccnt 0" ::: "memory");
}

// Build a 16-bit A/B fragment per the ISA 16x32 layout:
// lanes 0-15 hold K {0..7, 16..23}, lanes 16-31 hold K {8..15, 24..31}.
__device__ __forceinline__ v16h frag16(const _Float16* rowbase, int lh) {
  v8h lo = *(const v8h*)(rowbase + lh);
  v8h hi = *(const v8h*)(rowbase + lh + 16);
  return __builtin_shufflevector(lo, hi, 0,1,2,3,4,5,6,7,8,9,10,11,12,13,14,15);
}

// ---------------- elementwise helpers ----------------
__global__ __launch_bounds__(256) void cvt_f32_f16(const float* __restrict__ s,
                                                   _Float16* __restrict__ d, size_t n) {
  const size_t i = ((size_t)blockIdx.x * 256 + threadIdx.x) * 4;
  if (i >= n) return;
  const float4 v = *(const float4*)(s + i);
  v4h o = { (_Float16)v.x, (_Float16)v.y, (_Float16)v.z, (_Float16)v.w };
  *(v4h*)(d + i) = o;
}

__global__ void zero_f32(float* __restrict__ p, int n) {
  const int i = blockIdx.x * 256 + threadIdx.x;
  if (i < n) p[i] = 0.f;
}

// ---------------- projection / output GEMM: C = A(MxK,f16) * W(NxK,f16)^T + bias ----------------
// 256 threads = 8 wave32s in a 2(M) x 4(N) grid; 128x128 tile, BK=32 double-buffered
// via global_load_async_to_lds_b128. Each wave: 4x2 tiles of v_wmma_f32_16x16x32_f16.
template <bool OUTF16>
__global__ __launch_bounds__(256) void gemm_xwT(
    const _Float16* __restrict__ A, const _Float16* __restrict__ W,
    const float* __restrict__ bias, _Float16* __restrict__ outH,
    float* __restrict__ outF) {
  constexpr int K = DMOD, N = DMOD;
  __shared__ _Float16 sA[2][128 * LDSW];
  __shared__ _Float16 sB[2][128 * LDSW];
  const int tid = threadIdx.x;
  const _Float16* Ab = A + (size_t)blockIdx.y * 128 * K;
  const _Float16* Wb = W + (size_t)blockIdx.x * 128 * K;
  const int r0 = tid >> 2, c0 = (tid & 3) * 8;   // 16B chunk per lane, coalesced

  auto issue = [&](int buf, int kb) {
    const _Float16* ga = Ab + (size_t)r0 * K + kb + c0;
    const _Float16* gb = Wb + (size_t)r0 * K + kb + c0;
    async_copy_b128(ga,          &sA[buf][r0 * LDSW + c0]);
    async_copy_b128(ga + 64 * K, &sA[buf][(r0 + 64) * LDSW + c0]);
    async_copy_b128(gb,          &sB[buf][r0 * LDSW + c0]);
    async_copy_b128(gb + 64 * K, &sB[buf][(r0 + 64) * LDSW + c0]);
  };
  issue(0, 0);

  v8f zero = {};
  v8f acc[4][2];
  #pragma unroll
  for (int i = 0; i < 4; ++i)
    #pragma unroll
    for (int j = 0; j < 2; ++j) acc[i][j] = zero;

  const int wave = tid >> 5, lane = tid & 31;
  const int wm = (wave & 1) * 64, wn = (wave >> 1) * 32;
  const int lr = lane & 15, lh = (lane >> 4) * 8;

  for (int kb = 0; kb < K; kb += BK) {
    const int buf = (kb >> 5) & 1;
    wait_async0();
    __syncthreads();                       // tile 'buf' complete in LDS, all waves past 'buf^1' reads
    if (kb + BK < K) issue(buf ^ 1, kb + BK);
    v16h aF[4], bF[2];
    #pragma unroll
    for (int i = 0; i < 4; ++i) aF[i] = frag16(&sA[buf][(wm + i * 16 + lr) * LDSW], lh);
    #pragma unroll
    for (int j = 0; j < 2; ++j) bF[j] = frag16(&sB[buf][(wn + j * 16 + lr) * LDSW], lh);
    #pragma unroll
    for (int i = 0; i < 4; ++i)
      #pragma unroll
      for (int j = 0; j < 2; ++j)
        acc[i][j] = __builtin_amdgcn_wmma_f32_16x16x32_f16(
            false, aF[i], false, bF[j], (short)0, acc[i][j], false, false);
  }

  // C layout: lanes 0-15 VGPR g -> (M=g, N=lane); lanes 16-31 -> (M=8+g, N=lane-16)
  const int rowB = blockIdx.y * 128 + wm + (lane >> 4) * 8;
  const int colB = blockIdx.x * 128 + wn + (lane & 15);
  #pragma unroll
  for (int i = 0; i < 4; ++i)
    #pragma unroll
    for (int j = 0; j < 2; ++j) {
      const int col = colB + j * 16;
      const float bb = bias[col];
      #pragma unroll
      for (int g = 0; g < 8; ++g) {
        const size_t o = (size_t)(rowB + i * 16 + g) * N + col;
        const float v = acc[i][j][g] + bb;
        if (OUTF16) outH[o] = (_Float16)v; else outF[o] = v;
      }
    }
}

// ---------------- Gram GEMM with fused circular-diagonal reduction ----------------
// S_b = Q_b K_b^T per batch; mean_corr[b,tau] += (1/D) * sum over diagonal (i-j)==tau (mod L).
__global__ __launch_bounds__(256) void gram_diag(
    const _Float16* __restrict__ Q, const _Float16* __restrict__ Kt,
    float* __restrict__ mean_corr) {
  constexpr int K = DMOD;
  __shared__ _Float16 sA[2][128 * LDSW];
  __shared__ _Float16 sB[2][128 * LDSW];
  __shared__ float diag[256];
  const int tid = threadIdx.x;
  const int b = blockIdx.z;
  const _Float16* Ab = Q  + ((size_t)b * LSEQ + (size_t)blockIdx.y * 128) * K;
  const _Float16* Wb = Kt + ((size_t)b * LSEQ + (size_t)blockIdx.x * 128) * K;
  const int r0 = tid >> 2, c0 = (tid & 3) * 8;

  auto issue = [&](int buf, int kb) {
    const _Float16* ga = Ab + (size_t)r0 * K + kb + c0;
    const _Float16* gb = Wb + (size_t)r0 * K + kb + c0;
    async_copy_b128(ga,          &sA[buf][r0 * LDSW + c0]);
    async_copy_b128(ga + 64 * K, &sA[buf][(r0 + 64) * LDSW + c0]);
    async_copy_b128(gb,          &sB[buf][r0 * LDSW + c0]);
    async_copy_b128(gb + 64 * K, &sB[buf][(r0 + 64) * LDSW + c0]);
  };
  issue(0, 0);

  v8f zero = {};
  v8f acc[4][2];
  #pragma unroll
  for (int i = 0; i < 4; ++i)
    #pragma unroll
    for (int j = 0; j < 2; ++j) acc[i][j] = zero;

  const int wave = tid >> 5, lane = tid & 31;
  const int wm = (wave & 1) * 64, wn = (wave >> 1) * 32;
  const int lr = lane & 15, lh = (lane >> 4) * 8;

  for (int kb = 0; kb < K; kb += BK) {
    const int buf = (kb >> 5) & 1;
    wait_async0();
    __syncthreads();
    if (kb + BK < K) issue(buf ^ 1, kb + BK);
    v16h aF[4], bF[2];
    #pragma unroll
    for (int i = 0; i < 4; ++i) aF[i] = frag16(&sA[buf][(wm + i * 16 + lr) * LDSW], lh);
    #pragma unroll
    for (int j = 0; j < 2; ++j) bF[j] = frag16(&sB[buf][(wn + j * 16 + lr) * LDSW], lh);
    #pragma unroll
    for (int i = 0; i < 4; ++i)
      #pragma unroll
      for (int j = 0; j < 2; ++j)
        acc[i][j] = __builtin_amdgcn_wmma_f32_16x16x32_f16(
            false, aF[i], false, bF[j], (short)0, acc[i][j], false, false);
  }

  if (tid < 255) diag[tid] = 0.f;
  __syncthreads();
  #pragma unroll
  for (int i = 0; i < 4; ++i)
    #pragma unroll
    for (int j = 0; j < 2; ++j)
      #pragma unroll
      for (int g = 0; g < 8; ++g) {
        const int mi = wm + i * 16 + (lane >> 4) * 8 + g;   // 0..127
        const int nj = wn + j * 16 + (lane & 15);           // 0..127
        atomicAdd(&diag[mi - nj + 127], acc[i][j][g]);      // ds_add_f32
      }
  __syncthreads();
  if (tid < 255) {
    const int tau = ((int)blockIdx.y * 128 - (int)blockIdx.x * 128 + (tid - 127)
                     + 2 * LSEQ) & LMASK;
    atomicAdd(&mean_corr[(size_t)b * LSEQ + tau], diag[tid] * (1.0f / (float)DMOD));
  }
}

// ---------------- top-k over batch-mean correlation ----------------
__global__ void topk_kernel(const float* __restrict__ mc, int* __restrict__ idx) {
  __shared__ float g[LSEQ];
  __shared__ float rv[256];
  __shared__ int   ri[256];
  const int tid = threadIdx.x;
  for (int t = tid; t < LSEQ; t += 256) {
    float s = 0.f;
    for (int b = 0; b < BSZ; ++b) s += mc[(size_t)b * LSEQ + t];
    g[t] = s;  // argmax of sum == argmax of mean
  }
  __syncthreads();
  for (int k = 0; k < TOPK; ++k) {
    float best = -3.4e38f; int bi = 0;
    for (int t = tid; t < LSEQ; t += 256)
      if (g[t] > best) { best = g[t]; bi = t; }
    rv[tid] = best; ri[tid] = bi;
    __syncthreads();
    for (int s = 128; s > 0; s >>= 1) {
      if (tid < s && rv[tid + s] > rv[tid]) { rv[tid] = rv[tid + s]; ri[tid] = ri[tid + s]; }
      __syncthreads();
    }
    if (tid == 0) { idx[k] = ri[0]; g[ri[0]] = -3.4e38f; }
    __syncthreads();
  }
}

__global__ void softmax_w(const float* __restrict__ mc, const int* __restrict__ idx,
                          float* __restrict__ w) {
  const int b = threadIdx.x;
  if (b >= BSZ) return;
  float v[TOPK], mx = -3.4e38f;
  for (int i = 0; i < TOPK; ++i) { v[i] = mc[(size_t)b * LSEQ + idx[i]]; mx = fmaxf(mx, v[i]); }
  float s = 0.f;
  for (int i = 0; i < TOPK; ++i) { v[i] = __expf(v[i] - mx); s += v[i]; }
  for (int i = 0; i < TOPK; ++i) w[b * TOPK + i] = v[i] / s;
}

// ---------------- weighted circular roll-aggregate of V ----------------
// agg[b,l,d] = sum_k w[b,k] * V[b,(l+idx[k])%L, d]
__global__ __launch_bounds__(256) void roll_agg(
    const _Float16* __restrict__ V, const int* __restrict__ idx,
    const float* __restrict__ w, _Float16* __restrict__ agg) {
  const size_t e8 = ((size_t)blockIdx.x * 256 + threadIdx.x) * 8;
  const int d = (int)(e8 & (DMOD - 1));
  const size_t bl = e8 >> 10;            // D = 1024
  const int l = (int)(bl & LMASK);
  const int b = (int)(bl >> 11);         // L = 2048
  float acc[8];
  #pragma unroll
  for (int j = 0; j < 8; ++j) acc[j] = 0.f;
  #pragma unroll
  for (int k = 0; k < TOPK; ++k) {
    const int ls = (l + idx[k]) & LMASK;
    const v8h x = *(const v8h*)(V + (((size_t)b * LSEQ + ls) << 10) + d);
    const float wk = w[b * TOPK + k];
    #pragma unroll
    for (int j = 0; j < 8; ++j) acc[j] += wk * (float)x[j];
  }
  v8h o;
  #pragma unroll
  for (int j = 0; j < 8; ++j) o[j] = (_Float16)acc[j];
  *(v8h*)(agg + e8) = o;
}

// ---------------- driver ----------------
extern "C" void kernel_launch(void* const* d_in, const int* in_sizes, int n_in,
                              void* d_out, int out_size, void* d_ws, size_t ws_size,
                              hipStream_t stream) {
  (void)in_sizes; (void)n_in; (void)out_size; (void)ws_size;
  const float* queries = (const float*)d_in[0];
  const float* keys    = (const float*)d_in[1];
  const float* values  = (const float*)d_in[2];
  const float* Wq = (const float*)d_in[3];
  const float* bq = (const float*)d_in[4];
  const float* Wk = (const float*)d_in[5];
  const float* bk = (const float*)d_in[6];
  const float* Wv = (const float*)d_in[7];
  const float* bv = (const float*)d_in[8];
  const float* Wo = (const float*)d_in[9];
  const float* bo = (const float*)d_in[10];
  float* out = (float*)d_out;

  const size_t D2 = (size_t)DMOD * DMOD;          // 1 Mi elems
  const size_t XE = (size_t)BSZ * LSEQ * DMOD;    // 32 Mi elems
  char* ws = (char*)d_ws;
  size_t off = 0;
  _Float16* Wq16 = (_Float16*)(ws + off); off += D2 * 2;
  _Float16* Wk16 = (_Float16*)(ws + off); off += D2 * 2;
  _Float16* Wv16 = (_Float16*)(ws + off); off += D2 * 2;
  _Float16* Wo16 = (_Float16*)(ws + off); off += D2 * 2;
  _Float16* Xq16 = (_Float16*)(ws + off); off += XE * 2;
  _Float16* Xk16 = (_Float16*)(ws + off); off += XE * 2;
  _Float16* Xv16 = (_Float16*)(ws + off); off += XE * 2;
  _Float16* Q16  = (_Float16*)(ws + off); off += XE * 2;
  _Float16* K16  = (_Float16*)(ws + off); off += XE * 2;
  _Float16* V16  = (_Float16*)(ws + off); off += XE * 2;
  float*    mc   = (float*)(ws + off);    off += (size_t)BSZ * LSEQ * 4;
  int*      idx  = (int*)(ws + off);      off += 128;
  float*    tw   = (float*)(ws + off);    off += (size_t)BSZ * TOPK * 4;
  _Float16* AGG16 = Xq16;  // Xq16 dead after Q16 is produced

  const dim3 blk(256);
  // f32 -> f16 conversions (one pass; all subsequent tile loads are async B128)
  cvt_f32_f16<<<(unsigned)(XE / 1024), blk, 0, stream>>>(queries, Xq16, XE);
  cvt_f32_f16<<<(unsigned)(XE / 1024), blk, 0, stream>>>(keys,    Xk16, XE);
  cvt_f32_f16<<<(unsigned)(XE / 1024), blk, 0, stream>>>(values,  Xv16, XE);
  cvt_f32_f16<<<(unsigned)(D2 / 1024), blk, 0, stream>>>(Wq, Wq16, D2);
  cvt_f32_f16<<<(unsigned)(D2 / 1024), blk, 0, stream>>>(Wk, Wk16, D2);
  cvt_f32_f16<<<(unsigned)(D2 / 1024), blk, 0, stream>>>(Wv, Wv16, D2);
  cvt_f32_f16<<<(unsigned)(D2 / 1024), blk, 0, stream>>>(Wo, Wo16, D2);

  zero_f32<<<(BSZ * LSEQ) / 256, blk, 0, stream>>>(mc, BSZ * LSEQ);

  // Projections: q,k,v (WMMA f16 -> f32 acc -> f16 out)
  const dim3 gProj(DMOD / 128, (unsigned)(BSZ * LSEQ / 128));   // (8, 256)
  gemm_xwT<true><<<gProj, blk, 0, stream>>>(Xq16, Wq16, bq, Q16, nullptr);
  gemm_xwT<true><<<gProj, blk, 0, stream>>>(Xk16, Wk16, bk, K16, nullptr);
  gemm_xwT<true><<<gProj, blk, 0, stream>>>(Xv16, Wv16, bv, V16, nullptr);

  // Per-batch Gram Q K^T with fused circular-diagonal mean reduction
  const dim3 gGram(LSEQ / 128, LSEQ / 128, BSZ);                // (16, 16, 16)
  gram_diag<<<gGram, blk, 0, stream>>>(Q16, K16, mc);

  topk_kernel<<<1, 256, 0, stream>>>(mc, idx);
  softmax_w<<<1, 32, 0, stream>>>(mc, idx, tw);

  roll_agg<<<(unsigned)(XE / 2048), blk, 0, stream>>>(V16, idx, tw, AGG16);

  // Output projection straight to d_out (f32)
  gemm_xwT<false><<<gProj, blk, 0, stream>>>(AGG16, Wo16, bo, nullptr, out);
}
